// AOHA_62088047230998
// MI455X (gfx1250) — matrix operand order repaired and checked by reading, not compiled, not run on hardware
//
#include <hip/hip_runtime.h>
#include <hip/hip_bf16.h>

// ---------------------------------------------------------------------------
// Spiking attention block for MI455X (gfx1250, wave32, WMMA 16x16x32 f16,
// async global->LDS copies tracked by ASYNCcnt).
// B=8, T=4, L=512, D=512, H=8, HD=64.
// ---------------------------------------------------------------------------

typedef __attribute__((ext_vector_type(16))) _Float16 v16h;
typedef __attribute__((ext_vector_type(8)))  _Float16 v8h;
typedef __attribute__((ext_vector_type(8)))  float    v8f;

#define NB 8
#define NT 4
#define NL 512
#define ND 512
#define NH 8
#define NHD 64
#define NROWS (NB*NT*NL)       // 16384 rows for the projection GEMMs

// ---------------------------------------------------------------------------
// CDNA5 async copy: global memory -> LDS, 16 bytes per lane, ASYNCcnt-tracked.
// dsaddr = low 32 bits of the generic LDS pointer (flat aperture truncation).
// ---------------------------------------------------------------------------
__device__ inline void async_copy_b128(_Float16* lds_dst, const _Float16* gsrc) {
    unsigned int loff = (unsigned int)(unsigned long long)lds_dst;
    asm volatile("global_load_async_to_lds_b128 %0, %1, off"
                 :: "v"(loff), "v"(gsrc) : "memory");
}
__device__ inline void wait_async0() {
    asm volatile("s_wait_asynccnt 0x0" ::: "memory");
}
__device__ inline void wait_async4() {
    asm volatile("s_wait_asynccnt 0x4" ::: "memory");
}

// ---------------------------------------------------------------------------
// WMMA helpers: fragment loaders using contiguous k order (k = 16*(lane/16) + elem) for A and B.
// ---------------------------------------------------------------------------
__device__ inline v8f wmma_f16(v16h a, v16h b, v8f c) {
    // (neg_a, A, neg_b, B, c_mod, C, reuse_a, reuse_b)
    return __builtin_amdgcn_wmma_f32_16x16x32_f16(false, a, false, b,
                                                  (short)0, c, false, false);
}

// A-matrix 16x32 f16: lane (0-15) row M=lane, elems 0..15 -> K=0..15;
// lane (16-31) row M=lane-16, elems 0..15 -> K=16..31.
// Tile is row-major [M][K] with leading dim ld.
__device__ inline v16h load_a_frag(const _Float16* t, int ld, int mbase,
                                   int kbase, int lane) {
    int hf  = (lane >> 4) & 1;
    int row = mbase + (lane & 15);
    const _Float16* p = t + row * ld + kbase;
    v8h lo = *(const v8h*)(p + hf * 16);
    v8h hi = *(const v8h*)(p + hf * 16 + 8);
    v16h a;
#pragma unroll
    for (int i = 0; i < 8; i++) { a[i] = lo[i]; a[8 + i] = hi[i]; }
    return a;
}

// B-matrix 32x16 f16 where B[k][n] = t[n*ld + k]  (t holds rows = output
// columns, i.e. a weight matrix W[n][k] or a K-matrix tile K[m][d]).
// Lanes 0-15: K=0..15 across elems; lanes 16-31: K=16..31.
__device__ inline v16h load_b_tr(const _Float16* t, int ld, int nbase,
                                 int kbase, int lane) {
    int hf = (lane >> 4) & 1;
    int n  = nbase + (lane & 15);
    const _Float16* p = t + n * ld + kbase + hf * 16;
    v8h lo = *(const v8h*)(p);
    v8h hi = *(const v8h*)(p + 8);
    v16h b;
#pragma unroll
    for (int i = 0; i < 8; i++) { b[i] = lo[i]; b[8 + i] = hi[i]; }
    return b;
}

// B-matrix 32x16 f16 where B[k][n] = t[k*ld + n]  (row-major KxN tile, e.g. V).
__device__ inline v16h load_b_rm(const _Float16* t, int ld, int nbase,
                                 int kbase, int lane) {
    int hf = (lane >> 4) & 1;
    int n  = nbase + (lane & 15);
    v16h b;
#pragma unroll
    for (int i = 0; i < 16; i++)
        b[i] = t[(kbase + hf * 16 + i) * ld + n];
    return b;
}

// ---------------------------------------------------------------------------
// f32 -> f16 conversion
// ---------------------------------------------------------------------------
__global__ __launch_bounds__(256) void cvt_f32_f16(const float* __restrict__ s,
                                                   _Float16* __restrict__ d,
                                                   size_t n) {
    size_t i = (size_t)blockIdx.x * blockDim.x + threadIdx.x;
    size_t stride = (size_t)gridDim.x * blockDim.x;
    for (; i < n; i += stride) d[i] = (_Float16)s[i];
}

// ---------------------------------------------------------------------------
// Projection GEMM: out[row][n] = sum_k X[row][k] * W[n][k], fused BN epilogue.
// mode 0: Q path  -> fp32 pre-activation (LIF applied later)
// mode 1: K path  -> relu -> f16
// mode 2: V path  -> ternary {-1,0,1} -> f16
// mode 3: O path  -> +bias, BN -> fp32 pre-activation
// Tile: 128x128, K-step 32, 256 threads = 8 waves arranged 2 (M) x 4 (N).
// Double-buffered LDS; tiles filled by GLOBAL_LOAD_ASYNC_TO_LDS_B128 with a
// prefetch depth of 1 (4 async ops per wave per tile; wait asynccnt<=4 lets
// the current tile's ops retire while the next tile's stay in flight).
// ---------------------------------------------------------------------------
#define GLDA 40   // 32 K + 8 pad halves

__global__ __launch_bounds__(256) void gemm_bn_kernel(
    const _Float16* __restrict__ X, const _Float16* __restrict__ W,
    const float* __restrict__ g, const float* __restrict__ bb,
    const float* __restrict__ rm, const float* __restrict__ rv,
    const float* __restrict__ bias,
    float* __restrict__ outf, _Float16* __restrict__ outh, int mode) {

    __shared__ __align__(16) _Float16 As[2][128 * GLDA];
    __shared__ __align__(16) _Float16 Bs[2][128 * GLDA];

    const int tid = threadIdx.x, lane = tid & 31, wave = tid >> 5;
    const int wm = wave >> 2, wn = wave & 3;
    const size_t mtile = (size_t)blockIdx.x * 128;
    const int    ntile = blockIdx.y * 128;

    // Per-thread slots: 2 x 16B for A tile + 2 x 16B for B tile.
    auto issue_tile = [&](int buf, int k0) {
#pragma unroll
        for (int v = 0; v < 2; v++) {
            int id  = tid * 2 + v;       // 0..511 vec8 slots, 4 per row
            int row = id >> 2;
            int kc  = (id & 3) * 8;
            async_copy_b128(&As[buf][row * GLDA + kc],
                            &X[(mtile + row) * ND + k0 + kc]);
            async_copy_b128(&Bs[buf][row * GLDA + kc],
                            &W[(size_t)(ntile + row) * ND + k0 + kc]);
        }
    };

    v8f acc[4][2] = {};

    issue_tile(0, 0);
    for (int ks = 0; ks < ND / 32; ks++) {
        int buf = ks & 1;
        if (ks + 1 < ND / 32) {
            issue_tile(buf ^ 1, (ks + 1) * 32);  // prefetch next tile
            wait_async4();                       // current tile's 4 ops done
        } else {
            wait_async0();
        }
        __syncthreads();

        const _Float16* Ac = As[buf];
        const _Float16* Bc = Bs[buf];
        v16h bf0 = load_b_tr(Bc, GLDA, wn * 32, 0, lane);
        v16h bf1 = load_b_tr(Bc, GLDA, wn * 32 + 16, 0, lane);
#pragma unroll
        for (int fm = 0; fm < 4; fm++) {
            v16h a = load_a_frag(Ac, GLDA, wm * 64 + fm * 16, 0, lane);
            acc[fm][0] = wmma_f16(a, bf0, acc[fm][0]);
            acc[fm][1] = wmma_f16(a, bf1, acc[fm][1]);
        }
        __syncthreads();   // all waves done with buf before it is refilled
    }

    // Epilogue. C layout: elem r -> row = r + 8*(lane>>4), col = lane&15.
    const int nloc = lane & 15, hf = lane >> 4;
#pragma unroll
    for (int fn = 0; fn < 2; fn++) {
        int c = ntile + wn * 32 + fn * 16 + nloc;
        float sc = g[c] * rsqrtf(rv[c] + 1e-5f);
        float sh = bb[c] - rm[c] * sc;
        float bi = bias ? bias[c] : 0.f;
#pragma unroll
        for (int fm = 0; fm < 4; fm++) {
#pragma unroll
            for (int r = 0; r < 8; r++) {
                size_t row = mtile + wm * 64 + fm * 16 + hf * 8 + r;
                float y = (acc[fm][fn][r] + bi) * sc + sh;
                if (mode == 0 || mode == 3) {
                    outf[row * ND + c] = y;
                } else if (mode == 1) {
                    outh[row * ND + c] = (_Float16)fmaxf(y, 0.f);
                } else {
                    float tern = (y >= 1.f) ? 1.f : ((-y >= 1.f) ? -1.f : 0.f);
                    outh[row * ND + c] = (_Float16)tern;
                }
            }
        }
    }
}

// ---------------------------------------------------------------------------
// LIF temporal scan (snntorch Leaky, beta=0.5, thr=1, reset-by-subtraction).
// One thread per (b,l,d); T=4 sequential steps.
// mode 0: spikes -> f16 (Q path)   mode 1: spikes -> f32 (final output)
// ---------------------------------------------------------------------------
__global__ __launch_bounds__(256) void lif_kernel(const float* __restrict__ pre,
                                                  _Float16* __restrict__ outh,
                                                  float* __restrict__ outf,
                                                  int mode) {
    size_t i = (size_t)blockIdx.x * blockDim.x + threadIdx.x;
    const size_t n = (size_t)NB * NL * ND;
    if (i >= n) return;
    size_t b = i >> 18;                  // / (512*512)
    size_t r = i & ((1u << 18) - 1);
    float mem = 0.f;
#pragma unroll
    for (int t = 0; t < NT; t++) {
        size_t off = (((b * NT) + t) << 18) + r;
        float reset = (mem >= 1.f) ? 1.f : 0.f;       // from previous mem
        mem = 0.5f * mem + pre[off] - reset;          // THR == 1
        float spk = (mem >= 1.f) ? 1.f : 0.f;
        if (mode == 0) outh[off] = (_Float16)spk;
        else           outf[off] = spk;
    }
}

// ---------------------------------------------------------------------------
// Attention: per (b,t,h): S = Q(512x64) Kt(64x512) * 1/8 ; O = S V(512x64).
// Block = (bth, row-tile of 128). Q tile resident in LDS; loop over eight
// 64-row K/V tiles; S tile (128x64 f32) computed via WMMA, scaled, written
// f16 to LDS, then consumed as WMMA A against V.  S never touches HBM.
// K/V/Q tile fills use GLOBAL_LOAD_ASYNC_TO_LDS_B128.
// ---------------------------------------------------------------------------
#define ALD 72    // 64 + 8 pad halves

__global__ __launch_bounds__(256) void attn_kernel(
    const _Float16* __restrict__ Qp, const _Float16* __restrict__ Kp,
    const _Float16* __restrict__ Vp, _Float16* __restrict__ Op) {

    __shared__ __align__(16) _Float16 Qs[128 * ALD];
    __shared__ __align__(16) _Float16 Ks[64 * ALD];
    __shared__ __align__(16) _Float16 Vs[64 * ALD];
    __shared__ __align__(16) _Float16 Ss[128 * ALD];

    const int tid = threadIdx.x, lane = tid & 31, wave = tid >> 5;
    const int wm = wave >> 2, wn = wave & 3;      // 2 x 4 wave grid for S
    const int bt = blockIdx.x >> 3, h = blockIdx.x & 7;
    const int rt = blockIdx.y;                    // row tile (4 of 128)

    const _Float16* Qb = Qp + (((size_t)bt * NL + rt * 128) * ND) + h * NHD;
#pragma unroll
    for (int v = 0; v < 4; v++) {                 // 128x64 tile, 4 vec8/thread
        int id = tid * 4 + v;
        int row = id >> 3, c = (id & 7) * 8;
        async_copy_b128(&Qs[row * ALD + c], &Qb[(size_t)row * ND + c]);
    }

    v8f oacc[4] = {};                             // wave rows = wave*16..+16

    for (int j = 0; j < 8; j++) {
        const _Float16* Kb = Kp + (((size_t)bt * NL + j * 64) * ND) + h * NHD;
        const _Float16* Vb = Vp + (((size_t)bt * NL + j * 64) * ND) + h * NHD;
#pragma unroll
        for (int v = 0; v < 2; v++) {             // 64x64 tiles
            int id = tid * 2 + v;
            int row = id >> 3, c = (id & 7) * 8;
            async_copy_b128(&Ks[row * ALD + c], &Kb[(size_t)row * ND + c]);
            async_copy_b128(&Vs[row * ALD + c], &Vb[(size_t)row * ND + c]);
        }
        wait_async0();
        __syncthreads();

        // S tile 128x64: wave (wm,wn) owns rows wm*64+fm*16, cols wn*16.
        v8f sacc[4] = {};
        v16h b0 = load_b_tr(Ks, ALD, wn * 16, 0, lane);   // B[k][n]=K[n][k]
        v16h b1 = load_b_tr(Ks, ALD, wn * 16, 32, lane);
#pragma unroll
        for (int fm = 0; fm < 4; fm++) {
            v16h a0 = load_a_frag(Qs, ALD, wm * 64 + fm * 16, 0, lane);
            v16h a1 = load_a_frag(Qs, ALD, wm * 64 + fm * 16, 32, lane);
            sacc[fm] = wmma_f16(a0, b0, sacc[fm]);
            sacc[fm] = wmma_f16(a1, b1, sacc[fm]);
        }
        const int nloc = lane & 15, hf = lane >> 4;
#pragma unroll
        for (int fm = 0; fm < 4; fm++)
#pragma unroll
            for (int r = 0; r < 8; r++) {
                int row = wm * 64 + fm * 16 + hf * 8 + r;
                Ss[row * ALD + wn * 16 + nloc] =
                    (_Float16)(sacc[fm][r] * 0.125f);   // HD^-0.5
            }
        __syncthreads();

        // O += S(128x64) @ V(64x64): each wave owns 16 rows x 64 cols.
#pragma unroll
        for (int kk = 0; kk < 2; kk++) {
            v16h a = load_a_frag(Ss, ALD, wave * 16, kk * 32, lane);
#pragma unroll
            for (int fn = 0; fn < 4; fn++) {
                v16h b = load_b_rm(Vs, ALD, fn * 16, kk * 32, lane);
                oacc[fn] = wmma_f16(a, b, oacc[fn]);
            }
        }
        __syncthreads();   // Ks/Vs/Ss reads done before next j overwrites
    }

    const int nloc = lane & 15, hf = lane >> 4;
#pragma unroll
    for (int fn = 0; fn < 4; fn++)
#pragma unroll
        for (int r = 0; r < 8; r++) {
            size_t l = (size_t)rt * 128 + wave * 16 + hf * 8 + r;
            Op[(((size_t)bt * NL) + l) * ND + h * NHD + fn * 16 + nloc] =
                (_Float16)oacc[fn][r];
        }
}

// ---------------------------------------------------------------------------
// Host-side orchestration
// ---------------------------------------------------------------------------
extern "C" void kernel_launch(void* const* d_in, const int* in_sizes, int n_in,
                              void* d_out, int out_size, void* d_ws,
                              size_t ws_size, hipStream_t stream) {
    const float* x    = (const float*)d_in[0];
    const float* q_w  = (const float*)d_in[1];
    const float* q_g  = (const float*)d_in[2];
    const float* q_b  = (const float*)d_in[3];
    const float* q_rm = (const float*)d_in[4];
    const float* q_rv = (const float*)d_in[5];
    const float* k_w  = (const float*)d_in[6];
    const float* k_g  = (const float*)d_in[7];
    const float* k_b  = (const float*)d_in[8];
    const float* k_rm = (const float*)d_in[9];
    const float* k_rv = (const float*)d_in[10];
    const float* v_w  = (const float*)d_in[11];
    const float* v_g  = (const float*)d_in[12];
    const float* v_b  = (const float*)d_in[13];
    const float* v_rm = (const float*)d_in[14];
    const float* v_rv = (const float*)d_in[15];
    const float* o_w  = (const float*)d_in[16];
    const float* o_bi = (const float*)d_in[17];
    const float* o_g  = (const float*)d_in[18];
    const float* o_b  = (const float*)d_in[19];
    const float* o_rm = (const float*)d_in[20];
    const float* o_rv = (const float*)d_in[21];
    float* out = (float*)d_out;

    const size_t nx = (size_t)NB * NT * NL * ND;   // 8388608
    const size_t dd = (size_t)ND * ND;             // 262144

    char* w = (char*)d_ws;
    float*    pre   = (float*)w;     w += nx * sizeof(float);   // q_pre, then o_pre
    _Float16* xh    = (_Float16*)w;  w += nx * 2;
    _Float16* qwh   = (_Float16*)w;  w += dd * 2;
    _Float16* kwh   = (_Float16*)w;  w += dd * 2;
    _Float16* vwh   = (_Float16*)w;  w += dd * 2;
    _Float16* owh   = (_Float16*)w;  w += dd * 2;
    _Float16* qh    = (_Float16*)w;  w += nx * 2;
    _Float16* kh    = (_Float16*)w;  w += nx * 2;
    _Float16* vh    = (_Float16*)w;  w += nx * 2;
    _Float16* atth  = (_Float16*)w;  w += nx * 2;

    // 1) f16 conversions
    cvt_f32_f16<<<(int)((nx + 255) / 256), 256, 0, stream>>>(x, xh, nx);
    cvt_f32_f16<<<(int)((dd + 255) / 256), 256, 0, stream>>>(q_w, qwh, dd);
    cvt_f32_f16<<<(int)((dd + 255) / 256), 256, 0, stream>>>(k_w, kwh, dd);
    cvt_f32_f16<<<(int)((dd + 255) / 256), 256, 0, stream>>>(v_w, vwh, dd);
    cvt_f32_f16<<<(int)((dd + 255) / 256), 256, 0, stream>>>(o_w, owh, dd);

    dim3 ggrid(NROWS / 128, ND / 128);             // 128 x 4

    // 2) Q/K/V projections (WMMA GEMM + BN + activation epilogues)
    gemm_bn_kernel<<<ggrid, 256, 0, stream>>>(xh, qwh, q_g, q_b, q_rm, q_rv,
                                              nullptr, pre, nullptr, 0);
    gemm_bn_kernel<<<ggrid, 256, 0, stream>>>(xh, kwh, k_g, k_b, k_rm, k_rv,
                                              nullptr, nullptr, kh, 1);
    gemm_bn_kernel<<<ggrid, 256, 0, stream>>>(xh, vwh, v_g, v_b, v_rm, v_rv,
                                              nullptr, nullptr, vh, 2);

    // 3) LIF scan over T for Q -> binary spikes (f16)
    const size_t nlif = (size_t)NB * NL * ND;
    lif_kernel<<<(int)((nlif + 255) / 256), 256, 0, stream>>>(pre, qh, nullptr, 0);

    // 4) Attention (flash-style, S kept on-chip)
    attn_kernel<<<dim3(NB * NT * NH, NL / 128), 256, 0, stream>>>(qh, kh, vh, atth);

    // 5) Output projection + bias + BN -> fp32 pre-activation (reuse `pre`)
    gemm_bn_kernel<<<ggrid, 256, 0, stream>>>(atth, owh, o_g, o_b, o_rm, o_rv,
                                              o_bi, pre, nullptr, 3);

    // 6) Final LIF -> fp32 spikes into d_out
    lif_kernel<<<(int)((nlif + 255) / 256), 256, 0, stream>>>(pre, nullptr, out, 1);
}
